// block_tf_70592082477285
// MI455X (gfx1250) — matrix operand order repaired and checked by reading, not compiled, not run on hardware
//
#include <hip/hip_runtime.h>

// ---------------------------------------------------------------------------
// Sizes
// ---------------------------------------------------------------------------
#define NDRUG   1373
#define NMICRO  4096
#define NTOT    5469          // NDRUG + NMICRO
#define DMODEL  256
#define NHEAD   4
#define DHEAD   64
#define SCALE   0.125f        // 1/sqrt(64)
#define OFFB    1376          // 32-aligned start of "b" keys in concat K/V
#define NKPAD   5472          // OFFB + NMICRO
#define Q1PAD   1376          // drug query rows padded to 16
#define LNEPS   1e-5f

typedef __attribute__((ext_vector_type(16))) __bf16 v16bf;
typedef __attribute__((ext_vector_type(8)))  float  v8f;

union BfFrag { v16bf v; unsigned short u[16]; unsigned int d[8]; };

static __device__ __forceinline__ unsigned short f2bf(float f) {
    unsigned u = __float_as_uint(f);
    u += 0x7FFFu + ((u >> 16) & 1u);          // round-to-nearest-even
    return (unsigned short)(u >> 16);
}

// A-fragment (16x32 bf16, M x K), lane half h holds k = 16h + s (contiguous order). base -> element [row0][0].
static __device__ __forceinline__ v16bf load_a_frag(const unsigned short* base,
                                                    int ld, int k0) {
    int lane = threadIdx.x & 31;
    int sub  = lane >> 4;          // half of wave
    int lm   = lane & 15;          // row within tile
    const unsigned short* r = base + (long)lm * ld;
    BfFrag f;
#pragma unroll
    for (int v = 0; v < 8; ++v)
        f.d[v]     = *(const unsigned int*)(r + k0 + sub * 16 + 2 * v);
    return f.v;
}

// B-fragment (32x16 bf16, K x N) from "Bt" storage: Bt[n][k] contiguous in k.
// base -> element [n0][0].
static __device__ __forceinline__ v16bf load_b_frag(const unsigned short* base,
                                                    int ld, int k0) {
    int lane = threadIdx.x & 31;
    int sub  = lane >> 4;
    int lm   = lane & 15;          // column within tile
    const unsigned short* r = base + (long)lm * ld + k0 + sub * 16;
    BfFrag f;
#pragma unroll
    for (int v = 0; v < 8; ++v)
        f.d[v] = *(const unsigned int*)(r + 2 * v);
    return f.v;
}

static __device__ __forceinline__ v8f wmma_bf(v16bf a, v16bf b, v8f c) {
    return __builtin_amdgcn_wmma_f32_16x16x32_bf16(false, a, false, b,
                                                   (short)0, c, false, false);
}

// ---------------------------------------------------------------------------
// Small prep kernels
// ---------------------------------------------------------------------------
__global__ __launch_bounds__(256) void k_f2bf(const float* in,
                                              unsigned short* out, int n) {
    int i = blockIdx.x * 256 + threadIdx.x;
    if (i < n) out[i] = f2bf(in[i]);
}

// Wt[n][k] = bf16(W[k][n]) for a 256x256 (sub)matrix, W row-stride 256.
__global__ __launch_bounds__(256) void k_transpose(const float* W,
                                                   unsigned short* out) {
    int i = blockIdx.x * 256 + threadIdx.x;     // 65536 elements
    int n = i >> 8, k = i & 255;
    out[i] = f2bf(W[k * 256 + n]);
}

// beff[c][n] = b[n] + sum_k e[k] * W[(256+k)*256 + n]  for 12 (W,b,edge) combos
__global__ __launch_bounds__(256) void k_beff(
    const float* Wq1, const float* Wk1, const float* Wv1,
    const float* Wq2, const float* Wk2, const float* Wv2,
    const float* bq1, const float* bk1, const float* bv1,
    const float* bq2, const float* bk2, const float* bv2,
    const float* edge, float* beff) {
    const float* Ws[6] = {Wq1, Wk1, Wv1, Wq2, Wk2, Wv2};
    const float* bs[6] = {bq1, bk1, bv1, bq2, bk2, bv2};
    const int wsel[12] = {0, 1, 2, 0, 4, 5, 3, 1, 2, 3, 4, 5};
    const int esel[12] = {0, 0, 0, 2, 2, 2, 1, 1, 1, 3, 3, 3};
    int c = blockIdx.x, n = threadIdx.x;
    const float* W = Ws[wsel[c]];
    const float* e = edge + esel[c] * 256;
    float s = bs[wsel[c]][n];
    for (int k = 0; k < 256; ++k) s += e[k] * W[(256 + k) * 256 + n];
    beff[c * 256 + n] = s;
}

// ---------------------------------------------------------------------------
// Projection GEMM, 16x64 output tile per wave (A-fragment reused across 4
// column tiles; 4 independent accumulators -> back-to-back WMMA issue).
// out = bf16( A(Mx256) @ W(256x256) + beff ), scattered into head layouts.
//   mode 0: Q/K  out[h][rowOff+row][d]   (pad rows)
//   mode 1: Vt   out[h][d][rowOff+row]
// ---------------------------------------------------------------------------
__global__ __launch_bounds__(256) void k_proj(const unsigned short* A, int M,
                                              const unsigned short* Wt,
                                              const float* beff,
                                              unsigned short* out, int mode,
                                              int rowOff, int pad) {
    int wid   = threadIdx.x >> 5;
    int tiles = ((M + 15) >> 4) * 4;            // Mtiles * 4 col-groups of 64
    int tile  = blockIdx.x * 8 + wid;
    if (tile >= tiles) return;
    int tm = tile >> 2, cg = tile & 3;
    int lane = threadIdx.x & 31, sub = lane >> 4, lm = lane & 15;

    const unsigned short* Ab = A + (long)tm * 16 * 256;
    v8f acc[4];
    v8f zero = {};
#pragma unroll
    for (int c = 0; c < 4; ++c) acc[c] = zero;
#pragma unroll
    for (int k0 = 0; k0 < 256; k0 += 32) {
        v16bf a = load_a_frag(Ab, 256, k0);
#pragma unroll
        for (int c = 0; c < 4; ++c) {
            const unsigned short* Bb = Wt + (long)(cg * 64 + c * 16) * 256;
            acc[c] = wmma_bf(a, load_b_frag(Bb, 256, k0), acc[c]);
        }
    }

#pragma unroll
    for (int c = 0; c < 4; ++c) {
        int col = cg * 64 + c * 16 + lm;
        int h = col >> 6, d = col & 63;
        float bv = beff[col];
#pragma unroll
        for (int v = 0; v < 8; ++v) {
            int row = tm * 16 + sub * 8 + v;
            if (row < M) {
                unsigned short o = f2bf(acc[c][v] + bv);
                if (mode == 0)
                    out[(long)h * pad * 64 + (long)(rowOff + row) * 64 + d] = o;
                else
                    out[((long)h * 64 + d) * pad + rowOff + row] = o;
            }
        }
    }
}

// ---------------------------------------------------------------------------
// Generic GEMM, 16x64 tile per wave:
// C(Mx256) = A_bf(Mx256) @ W_bf(256x256) [+bias][+residual][relu]
// ---------------------------------------------------------------------------
__global__ __launch_bounds__(256) void k_gemm(const unsigned short* A, int M,
                                              const unsigned short* Wt,
                                              const float* bias,
                                              const float* residual, int relu,
                                              float* outF,
                                              unsigned short* outBf) {
    int wid   = threadIdx.x >> 5;
    int tiles = ((M + 15) >> 4) * 4;
    int tile  = blockIdx.x * 8 + wid;
    if (tile >= tiles) return;
    int tm = tile >> 2, cg = tile & 3;
    int lane = threadIdx.x & 31, sub = lane >> 4, lm = lane & 15;

    const unsigned short* Ab = A + (long)tm * 16 * 256;
    v8f acc[4];
    v8f zero = {};
#pragma unroll
    for (int c = 0; c < 4; ++c) acc[c] = zero;
#pragma unroll
    for (int k0 = 0; k0 < 256; k0 += 32) {
        v16bf a = load_a_frag(Ab, 256, k0);
#pragma unroll
        for (int c = 0; c < 4; ++c) {
            const unsigned short* Bb = Wt + (long)(cg * 64 + c * 16) * 256;
            acc[c] = wmma_bf(a, load_b_frag(Bb, 256, k0), acc[c]);
        }
    }

#pragma unroll
    for (int c = 0; c < 4; ++c) {
        int col = cg * 64 + c * 16 + lm;
        float bv = bias ? bias[col] : 0.f;
#pragma unroll
        for (int v = 0; v < 8; ++v) {
            int row = tm * 16 + sub * 8 + v;
            if (row < M) {
                float val = acc[c][v] + bv;
                if (residual) val += residual[(long)row * 256 + col];
                if (relu) val = fmaxf(val, 0.f);
                if (outF)  outF [(long)row * 256 + col] = val;
                if (outBf) outBf[(long)row * 256 + col] = f2bf(val);
            }
        }
    }
}

// ---------------------------------------------------------------------------
// Flash attention.  One wave = one (16-query-row, head) tile.
// Keys [0,nA) scored with Qa, keys [offB, nkTot) scored with Qb (two phases).
// K: [h][key][d] bf16 (pad nkTot), Vt: [h][d][key] bf16.
// ---------------------------------------------------------------------------
__global__ __launch_bounds__(256) void k_flash(
    const unsigned short* Qa, const unsigned short* Qb, int qPad,
    const unsigned short* K, const unsigned short* Vt, int nq, int nA,
    int offB, int nkTot, float* mess, unsigned short* messBf) {
    __shared__ unsigned short plds[8 * 512];    // 8 waves x (16x32 bf16)

    int wid     = threadIdx.x >> 5;
    int nqTiles = (nq + 15) >> 4;
    int qtile   = blockIdx.x * 8 + wid;
    if (qtile >= nqTiles) return;
    int h  = blockIdx.y;
    int q0 = qtile * 16;
    int lane = threadIdx.x & 31, sub = lane >> 4, lm = lane & 15;
    unsigned short* myl = plds + wid * 512;

    const unsigned short* Kh = K  + (long)h * nkTot * 64;
    const unsigned short* Vh = Vt + (long)h * 64 * nkTot;

    float m8[8], l8[8];
    v8f O[4];
    v8f zero = {};
#pragma unroll
    for (int v = 0; v < 8; ++v) { m8[v] = -3e38f; l8[v] = 0.f; }
#pragma unroll
    for (int t = 0; t < 4; ++t) O[t] = zero;

    for (int phase = 0; phase < 2; ++phase) {
        const unsigned short* Qbuf = phase ? Qb : Qa;
        const unsigned short* Qrow = Qbuf + (long)h * qPad * 64 + (long)q0 * 64;
        v16bf qf0 = load_a_frag(Qrow, 64, 0);
        v16bf qf1 = load_a_frag(Qrow, 64, 32);
        int kbS   = phase ? offB : 0;
        int kbE   = phase ? nkTot : offB;
        int limit = phase ? nkTot : nA;

        for (int kb = kbS; kb < kbE; kb += 32) {
            if (kb + 32 < kbE)
                __builtin_prefetch(Kh + (long)(kb + 32) * 64, 0, 1);

            const unsigned short* kcol = Kh + (long)kb * 64;
            v8f S0 = zero, S1 = zero;
            S0 = wmma_bf(qf0, load_b_frag(kcol, 64, 0),  S0);
            S0 = wmma_bf(qf1, load_b_frag(kcol, 64, 32), S0);
            S1 = wmma_bf(qf0, load_b_frag(kcol + 16 * 64, 64, 0),  S1);
            S1 = wmma_bf(qf1, load_b_frag(kcol + 16 * 64, 64, 32), S1);

            int k0g = kb + lm, k1g = kb + 16 + lm;
            float s0a[8], s1a[8], alpha[8];
#pragma unroll
            for (int v = 0; v < 8; ++v) {
                s0a[v] = (k0g < limit) ? S0[v] * SCALE : -1e30f;
                s1a[v] = (k1g < limit) ? S1[v] * SCALE : -1e30f;
            }
#pragma unroll
            for (int v = 0; v < 8; ++v) {
                float t = fmaxf(s0a[v], s1a[v]);
#pragma unroll
                for (int off = 1; off < 16; off <<= 1)
                    t = fmaxf(t, __shfl_xor(t, off, 32));
                float mn = fmaxf(m8[v], t);
                alpha[v] = __expf(m8[v] - mn);
                m8[v]    = mn;
                s0a[v]   = __expf(s0a[v] - mn);
                s1a[v]   = __expf(s1a[v] - mn);
                float sm = s0a[v] + s1a[v];
#pragma unroll
                for (int off = 1; off < 16; off <<= 1)
                    sm += __shfl_xor(sm, off, 32);
                l8[v] = l8[v] * alpha[v] + sm;
            }
#pragma unroll
            for (int t = 0; t < 4; ++t)
#pragma unroll
                for (int v = 0; v < 8; ++v) O[t][v] *= alpha[v];

            // P tile (16x32) -> LDS (bf16), then reload as A-fragment.
#pragma unroll
            for (int v = 0; v < 8; ++v) {
                myl[(sub * 8 + v) * 32 + lm]      = f2bf(s0a[v]);
                myl[(sub * 8 + v) * 32 + 16 + lm] = f2bf(s1a[v]);
            }
            asm volatile("s_wait_dscnt 0" ::: "memory");
            v16bf pf = load_a_frag(myl, 32, 0);

#pragma unroll
            for (int t = 0; t < 4; ++t) {
                v16bf vb = load_b_frag(Vh + (long)(t * 16) * nkTot, nkTot, kb);
                O[t] = wmma_bf(pf, vb, O[t]);
            }
        }
    }

#pragma unroll
    for (int t = 0; t < 4; ++t)
#pragma unroll
        for (int v = 0; v < 8; ++v) {
            int row = q0 + sub * 8 + v;
            if (row < nq) {
                int col = h * 64 + t * 16 + lm;
                float val = O[t][v] / l8[v];
                mess  [(long)row * 256 + col] = val;
                messBf[(long)row * 256 + col] = f2bf(val);
            }
        }
}

// ---------------------------------------------------------------------------
// Row LayerNorm over 256 columns, one block per row.
// ---------------------------------------------------------------------------
__global__ __launch_bounds__(256) void k_ln(const float* in, const float* g,
                                            const float* b, float* outF,
                                            unsigned short* outBf) {
    __shared__ float red[256];
    int row = blockIdx.x, tid = threadIdx.x;
    float x = in[(long)row * 256 + tid];
    red[tid] = x;
    __syncthreads();
    for (int s = 128; s > 0; s >>= 1) {
        if (tid < s) red[tid] += red[tid + s];
        __syncthreads();
    }
    float mean = red[0] * (1.f / 256.f);
    __syncthreads();
    float d = x - mean;
    red[tid] = d * d;
    __syncthreads();
    for (int s = 128; s > 0; s >>= 1) {
        if (tid < s) red[tid] += red[tid + s];
        __syncthreads();
    }
    float var = red[0] * (1.f / 256.f);
    float y = d * rsqrtf(var + LNEPS) * g[tid] + b[tid];
    outF[(long)row * 256 + tid] = y;
    if (outBf) outBf[(long)row * 256 + tid] = f2bf(y);
}

// ---------------------------------------------------------------------------
// Host orchestration
// ---------------------------------------------------------------------------
extern "C" void kernel_launch(void* const* d_in, const int* in_sizes, int n_in,
                              void* d_out, int out_size, void* d_ws,
                              size_t ws_size, hipStream_t stream) {
    (void)in_sizes; (void)n_in; (void)out_size; (void)ws_size;
    const float* node = (const float*)d_in[0];
    const float* edge = (const float*)d_in[1];
    const float* Wq1 = (const float*)d_in[2];  const float* bq1 = (const float*)d_in[3];
    const float* Wk1 = (const float*)d_in[4];  const float* bk1 = (const float*)d_in[5];
    const float* Wv1 = (const float*)d_in[6];  const float* bv1 = (const float*)d_in[7];
    const float* Wq2 = (const float*)d_in[8];  const float* bq2 = (const float*)d_in[9];
    const float* Wk2 = (const float*)d_in[10]; const float* bk2 = (const float*)d_in[11];
    const float* Wv2 = (const float*)d_in[12]; const float* bv2 = (const float*)d_in[13];
    const float* W1  = (const float*)d_in[14]; const float* b1  = (const float*)d_in[15];
    const float* W2  = (const float*)d_in[16]; const float* b2  = (const float*)d_in[17];
    const float* W3  = (const float*)d_in[18]; const float* b3  = (const float*)d_in[19];
    const float* lng = (const float*)d_in[20]; const float* lnb = (const float*)d_in[21];

    char* ws = (char*)d_ws;
    size_t off = 0;
    auto alloc = [&](size_t bytes) -> void* {
        void* p = ws + off;
        off += (bytes + 255) & ~(size_t)255;
        return p;
    };
    typedef unsigned short u16;

    u16* node_bf = (u16*)alloc((size_t)NTOT * 256 * 2);
    u16* wt[6];
    for (int i = 0; i < 6; ++i) wt[i] = (u16*)alloc(65536 * 2);
    u16* w123t[3];
    for (int i = 0; i < 3; ++i) w123t[i] = (u16*)alloc(65536 * 2);
    float* beff = (float*)alloc(12 * 256 * 4);

    u16* Qa1 = (u16*)alloc((size_t)NHEAD * Q1PAD * 64 * 2);
    u16* Qb1 = (u16*)alloc((size_t)NHEAD * Q1PAD * 64 * 2);
    u16* Qa2 = (u16*)alloc((size_t)NHEAD * NMICRO * 64 * 2);
    u16* Qb2 = (u16*)alloc((size_t)NHEAD * NMICRO * 64 * 2);

    size_t kvBytes = (size_t)NHEAD * NKPAD * 64 * 2;   // multiple of 256
    u16* K1 = (u16*)alloc(kvBytes);
    u16* V1 = (u16*)alloc(kvBytes);
    u16* K2 = (u16*)alloc(kvBytes);
    u16* V2 = (u16*)alloc(kvBytes);

    float* mess1   = (float*)alloc((size_t)Q1PAD * 256 * 4);
    u16*   mess1bf = (u16*)  alloc((size_t)Q1PAD * 256 * 2);
    float* mess2   = (float*)alloc((size_t)NMICRO * 256 * 4);
    u16*   mess2bf = (u16*)  alloc((size_t)NMICRO * 256 * 2);
    float* tbuf    = (float*)alloc((size_t)NMICRO * 256 * 4);
    float* cof1    = (float*)alloc((size_t)Q1PAD * 256 * 4);
    u16*   cof1bf  = (u16*)  alloc((size_t)Q1PAD * 256 * 2);
    float* cof2    = (float*)alloc((size_t)NMICRO * 256 * 4);
    u16*   cof2bf  = (u16*)  alloc((size_t)NMICRO * 256 * 2);
    u16*   h1bf    = (u16*)  alloc((size_t)Q1PAD * 256 * 2);
    u16*   h2bf    = (u16*)  alloc((size_t)NMICRO * 256 * 2);

    // Zero padded K/V (contiguous 4 buffers) so padded keys are benign.
    (void)hipMemsetAsync(K1, 0, kvBytes * 4, stream);

    // --- prep ---
    k_f2bf<<<(NTOT * 256 + 255) / 256, 256, 0, stream>>>(node, node_bf,
                                                         NTOT * 256);
    const float* Wsrc[6] = {Wq1, Wk1, Wv1, Wq2, Wk2, Wv2};
    for (int i = 0; i < 6; ++i)
        k_transpose<<<256, 256, 0, stream>>>(Wsrc[i], wt[i]);
    const float* Wffn[3] = {W1, W2, W3};
    for (int i = 0; i < 3; ++i)
        k_transpose<<<256, 256, 0, stream>>>(Wffn[i], w123t[i]);
    k_beff<<<12, 256, 0, stream>>>(Wq1, Wk1, Wv1, Wq2, Wk2, Wv2, bq1, bk1,
                                   bv1, bq2, bk2, bv2, edge, beff);

    // 16x64 tiles per wave: Mtiles*4 tiles, 8 waves per block
    auto gblk = [](int M) { return ((((M + 15) / 16) * 4) + 7) / 8; };
    const u16* drugbf  = node_bf;
    const u16* microbf = node_bf + (size_t)NDRUG * 256;

    // --- 12 projections ---
    k_proj<<<gblk(NDRUG), 256, 0, stream>>>(drugbf,  NDRUG,  wt[0], beff + 0 * 256,  Qa1, 0, 0,    Q1PAD);
    k_proj<<<gblk(NDRUG), 256, 0, stream>>>(drugbf,  NDRUG,  wt[1], beff + 1 * 256,  K1,  0, 0,    NKPAD);
    k_proj<<<gblk(NDRUG), 256, 0, stream>>>(drugbf,  NDRUG,  wt[2], beff + 2 * 256,  V1,  1, 0,    NKPAD);
    k_proj<<<gblk(NDRUG), 256, 0, stream>>>(drugbf,  NDRUG,  wt[0], beff + 3 * 256,  Qb1, 0, 0,    Q1PAD);
    k_proj<<<gblk(NMICRO), 256, 0, stream>>>(microbf, NMICRO, wt[4], beff + 4 * 256,  K1,  0, OFFB, NKPAD);
    k_proj<<<gblk(NMICRO), 256, 0, stream>>>(microbf, NMICRO, wt[5], beff + 5 * 256,  V1,  1, OFFB, NKPAD);
    k_proj<<<gblk(NMICRO), 256, 0, stream>>>(microbf, NMICRO, wt[3], beff + 6 * 256,  Qa2, 0, 0,    NMICRO);
    k_proj<<<gblk(NDRUG), 256, 0, stream>>>(drugbf,  NDRUG,  wt[1], beff + 7 * 256,  K2,  0, 0,    NKPAD);
    k_proj<<<gblk(NDRUG), 256, 0, stream>>>(drugbf,  NDRUG,  wt[2], beff + 8 * 256,  V2,  1, 0,    NKPAD);
    k_proj<<<gblk(NMICRO), 256, 0, stream>>>(microbf, NMICRO, wt[3], beff + 9 * 256,  Qb2, 0, 0,    NMICRO);
    k_proj<<<gblk(NMICRO), 256, 0, stream>>>(microbf, NMICRO, wt[4], beff + 10 * 256, K2,  0, OFFB, NKPAD);
    k_proj<<<gblk(NMICRO), 256, 0, stream>>>(microbf, NMICRO, wt[5], beff + 11 * 256, V2,  1, OFFB, NKPAD);

    // --- flash attention ---
    dim3 g1((((NDRUG + 15) / 16) + 7) / 8, NHEAD);
    k_flash<<<g1, 256, 0, stream>>>(Qa1, Qb1, Q1PAD, K1, V1, NDRUG, NDRUG,
                                    OFFB, NKPAD, mess1, mess1bf);
    dim3 g2(((NMICRO / 16) + 7) / 8, NHEAD);
    k_flash<<<g2, 256, 0, stream>>>(Qa2, Qb2, NMICRO, K2, V2, NMICRO, NDRUG,
                                    OFFB, NKPAD, mess2, mess2bf);

    float* outF = (float*)d_out;

    // --- FFN, drug ---
    k_gemm<<<gblk(NDRUG), 256, 0, stream>>>(mess1bf, NDRUG, w123t[0], b1,
                                            node, 0, tbuf, nullptr);
    k_ln<<<NDRUG, 256, 0, stream>>>(tbuf, lng, lnb, cof1, cof1bf);
    k_gemm<<<gblk(NDRUG), 256, 0, stream>>>(cof1bf, NDRUG, w123t[1], b2,
                                            nullptr, 1, nullptr, h1bf);
    k_gemm<<<gblk(NDRUG), 256, 0, stream>>>(h1bf, NDRUG, w123t[2], b3,
                                            cof1, 0, tbuf, nullptr);
    k_ln<<<NDRUG, 256, 0, stream>>>(tbuf, lng, lnb, outF, nullptr);

    // --- FFN, micro ---
    k_gemm<<<gblk(NMICRO), 256, 0, stream>>>(mess2bf, NMICRO, w123t[0], b1,
                                             node + (size_t)NDRUG * 256, 0,
                                             tbuf, nullptr);
    k_ln<<<NMICRO, 256, 0, stream>>>(tbuf, lng, lnb, cof2, cof2bf);
    k_gemm<<<gblk(NMICRO), 256, 0, stream>>>(cof2bf, NMICRO, w123t[1], b2,
                                             nullptr, 1, nullptr, h2bf);
    k_gemm<<<gblk(NMICRO), 256, 0, stream>>>(h2bf, NMICRO, w123t[2], b3,
                                             cof2, 0, tbuf, nullptr);
    k_ln<<<NMICRO, 256, 0, stream>>>(tbuf, lng, lnb,
                                     outF + (size_t)NDRUG * 256, nullptr);
}
